// SelfAttention_88940182765692
// MI455X (gfx1250) — matrix operand order repaired and checked by reading, not compiled, not run on hardware
//
#include <hip/hip_runtime.h>

// ---------------------------------------------------------------------------
// Self-attention, MI455X (gfx1250), f16 WMMA with f32 accumulation.
// Pipeline:
//   1) convert x + 4 weights fp32 -> f16 (workspace)
//   2) Q = x@wq.T, K = x@wk.T  stored [H][S][D] f16 ; V stored transposed [H][D][S]
//   3) flash-attention (online softmax, no 1/sqrt(D) scale per reference),
//      output stored [H][S][D] f16  == reference's buggy reshape(B,S,H*D) flat
//   4) out = attn_flat[4096x1024] @ wo.T + bo   (fp32 output)
// All GEMM / attention K-loops are explicitly double-buffered so global-load
// waits overlap the WMMA stream.
// ---------------------------------------------------------------------------

typedef __attribute__((ext_vector_type(16))) _Float16 v16h;
typedef __attribute__((ext_vector_type(8)))  _Float16 v8h;
typedef __attribute__((ext_vector_type(4)))  _Float16 v4h;
typedef __attribute__((ext_vector_type(8)))  float    v8f;

#define S_LEN 4096
#define E_DIM 1024
#define H_NUM 16
#define D_HEAD 64

static __device__ __forceinline__ v16h cat8(v8h lo, v8h hi) {
  return __builtin_shufflevector(lo, hi, 0,1,2,3,4,5,6,7,8,9,10,11,12,13,14,15);
}

static __device__ __forceinline__ v8f wmma_f16(v16h a, v16h b, v8f c) {
  return __builtin_amdgcn_wmma_f32_16x16x32_f16(false, a, false, b, (short)0, c,
                                                false, false);
}

// ---------------------------------------------------------------------------
// fp32 -> f16 conversion, 4 elements per thread
// ---------------------------------------------------------------------------
__global__ void __launch_bounds__(256)
f32_to_f16_kernel(const float* __restrict__ in, _Float16* __restrict__ out, int n) {
  int i = (blockIdx.x * blockDim.x + threadIdx.x) * 4;
  if (i < n) {
    float4 v = *(const float4*)(in + i);
    v4h o;
    o[0] = (_Float16)v.x; o[1] = (_Float16)v.y;
    o[2] = (_Float16)v.z; o[3] = (_Float16)v.w;
    *(v4h*)(out + i) = o;
  }
}

// ---------------------------------------------------------------------------
// GEMM helpers (A-frag: 16x32 f16, contiguous k = 16*lh + s; B-frag: 32x16 from W rows)
// ---------------------------------------------------------------------------
static __device__ __forceinline__ void load_gemm_frags(
    const _Float16* __restrict__ A, const _Float16* __restrict__ W,
    int m0, int n0, int k0, int ln, int lh, v16h (&af)[4], v16h (&bf)[4]) {
#pragma unroll
  for (int i = 0; i < 4; ++i) {
    const _Float16* ap = A + (m0 + i * 16 + ln) * 1024 + k0 + lh * 16;
    af[i] = *(const v16h*)ap;
  }
#pragma unroll
  for (int j = 0; j < 4; ++j)
    bf[j] = *(const v16h*)(W + (n0 + j * 16 + ln) * 1024 + k0 + lh * 16);
}

static __device__ __forceinline__ void mma16(const v16h (&af)[4],
                                             const v16h (&bf)[4],
                                             v8f (&acc)[16]) {
#pragma unroll
  for (int i = 0; i < 4; ++i)
#pragma unroll
    for (int j = 0; j < 4; ++j)
      acc[i * 4 + j] = wmma_f16(af[i], bf[j], acc[i * 4 + j]);
}

// ---------------------------------------------------------------------------
// C = A[4096x1024] @ W.T  (W row-major [N=1024 x K=1024])
// mode 0: f16 scatter [h][s][d] (Q/K); mode 1: f16 scatter [h][d][s] (Vt);
// mode 2: f32 linear + bias.
// 256 threads = 8 waves; wave tile 64x64; block tile 128(M) x 256(N).
// ---------------------------------------------------------------------------
__global__ void __launch_bounds__(256)
gemm_xwt_kernel(const _Float16* __restrict__ A, const _Float16* __restrict__ W,
                _Float16* __restrict__ outh, const float* __restrict__ bias,
                float* __restrict__ outf, int mode) {
  const int lane = threadIdx.x & 31;
  const int wv   = threadIdx.x >> 5;
  const int ln = lane & 15, lh = lane >> 4;
  const int m0 = blockIdx.y * 128 + (wv >> 2) * 64;
  const int n0 = blockIdx.x * 256 + (wv & 3) * 64;

  v8f acc[16];
  const v8f z8 = {0.f,0.f,0.f,0.f,0.f,0.f,0.f,0.f};
#pragma unroll
  for (int i = 0; i < 16; ++i) acc[i] = z8;

  v16h afA[4], bfA[4], afB[4], bfB[4];
  load_gemm_frags(A, W, m0, n0, 0, ln, lh, afA, bfA);

  // Double-buffered K loop: issue next fragments before consuming current.
  for (int k0 = 0; k0 < 1024; k0 += 64) {
    __builtin_prefetch(A + (m0 + ln) * 1024 + k0 + 128, 0, 1);
    __builtin_prefetch(W + (n0 + ln) * 1024 + k0 + 128, 0, 1);
    load_gemm_frags(A, W, m0, n0, k0 + 32, ln, lh, afB, bfB);
    mma16(afA, bfA, acc);
    const int kn = (k0 + 64 < 1024) ? k0 + 64 : 0;  // clamped redundant preload
    load_gemm_frags(A, W, m0, n0, kn, ln, lh, afA, bfA);
    mma16(afB, bfB, acc);
  }

  if (mode == 2) {
    float bj[4];
#pragma unroll
    for (int j = 0; j < 4; ++j) bj[j] = bias[n0 + j * 16 + ln];
#pragma unroll
    for (int i = 0; i < 4; ++i)
#pragma unroll
      for (int j = 0; j < 4; ++j)
#pragma unroll
        for (int r = 0; r < 8; ++r) {
          int m = m0 + i * 16 + r + lh * 8;
          int n = n0 + j * 16 + ln;
          outf[m * 1024 + n] = acc[i * 4 + j][r] + bj[j];
        }
  } else {
#pragma unroll
    for (int i = 0; i < 4; ++i)
#pragma unroll
      for (int j = 0; j < 4; ++j)
#pragma unroll
        for (int r = 0; r < 8; ++r) {
          int m = m0 + i * 16 + r + lh * 8;
          int n = n0 + j * 16 + ln;
          float v = acc[i * 4 + j][r];
          if (mode == 0)
            outh[((n >> 6) * S_LEN + m) * D_HEAD + (n & 63)] = (_Float16)v;
          else
            outh[(n >> 6) * (D_HEAD * S_LEN) + (n & 63) * S_LEN + m] = (_Float16)v;
        }
  }
}

// ---------------------------------------------------------------------------
// Flash attention helpers
// ---------------------------------------------------------------------------
static __device__ __forceinline__ void load_kfrags(
    const _Float16* __restrict__ Kh, int kb, int ln, int lh, v16h (&kf)[2][4]) {
#pragma unroll
  for (int s = 0; s < 2; ++s)
#pragma unroll
    for (int j = 0; j < 4; ++j)
      kf[s][j] = *(const v16h*)(Kh + (kb + j * 16 + ln) * D_HEAD + s * 32 + lh * 16);
}

// One 64-key block: QK^T (WMMA) -> online softmax -> P via LDS -> P@V (WMMA).
// Preloads K fragments for next_kb between the QK WMMAs and the softmax VALU.
static __device__ __forceinline__ void attn_block(
    int kb, int next_kb, const _Float16* __restrict__ Kh,
    const _Float16* __restrict__ Vh, _Float16* __restrict__ pw,
    int ln, int lh, const v16h (&qf)[2], v16h (&kfc)[2][4], v16h (&kfn)[2][4],
    v8f (&o)[4], float (&rm)[8], float (&rl)[8]) {
  const v8f z8 = {0.f,0.f,0.f,0.f,0.f,0.f,0.f,0.f};

  // Issue V fragment loads for this block first (consumed at the end).
  v16h vf[2][4];
#pragma unroll
  for (int s = 0; s < 2; ++s)
#pragma unroll
    for (int jd = 0; jd < 4; ++jd)
      vf[s][jd] =
          *(const v16h*)(Vh + (jd * 16 + ln) * S_LEN + kb + s * 32 + lh * 16);

  // ---- S-tile = Q K^T : 16 x 64, f32 ----
  v8f st[4];
#pragma unroll
  for (int j = 0; j < 4; ++j) st[j] = z8;
#pragma unroll
  for (int s = 0; s < 2; ++s)
#pragma unroll
    for (int j = 0; j < 4; ++j) st[j] = wmma_f16(qf[s], kfc[s][j], st[j]);

  // Preload next block's K fragments; latency hidden under softmax.
  load_kfrags(Kh, next_kb, ln, lh, kfn);

  // ---- online softmax (rows live across the two 16-lane halves) ----
  float sc[8];
#pragma unroll
  for (int r = 0; r < 8; ++r) {
    float mx = fmaxf(fmaxf(st[0][r], st[1][r]), fmaxf(st[2][r], st[3][r]));
#pragma unroll
    for (int off = 8; off >= 1; off >>= 1) mx = fmaxf(mx, __shfl_xor(mx, off, 16));
    float nm = fmaxf(rm[r], mx);
    sc[r] = __expf(rm[r] - nm);
    rm[r] = nm;
    float sum = 0.f;
#pragma unroll
    for (int j = 0; j < 4; ++j) {
      float p = __expf(st[j][r] - nm);
      st[j][r] = p;
      sum += p;
    }
#pragma unroll
    for (int off = 8; off >= 1; off >>= 1) sum += __shfl_xor(sum, off, 16);
    rl[r] = rl[r] * sc[r] + sum;
  }
#pragma unroll
  for (int j = 0; j < 4; ++j)
#pragma unroll
    for (int r = 0; r < 8; ++r) o[j][r] *= sc[r];

  // ---- P: C layout -> A layout via wave-private LDS (in-order per wave) ----
#pragma unroll
  for (int j = 0; j < 4; ++j)
#pragma unroll
    for (int r = 0; r < 8; ++r)
      pw[(r + lh * 8) * 64 + j * 16 + ln] = (_Float16)st[j][r];
  // Explicit CDNA5 split-counter fence: drain DS stores before DS reloads.
  asm volatile("s_wait_dscnt 0x0" ::: "memory");
  v16h pf[2];
  const _Float16* prow = pw + ln * 64;
#pragma unroll
  for (int s = 0; s < 2; ++s) {
    pf[s] = *(const v16h*)(prow + s * 32 + lh * 16);
  }

  // ---- O += P @ V ----
#pragma unroll
  for (int s = 0; s < 2; ++s)
#pragma unroll
    for (int jd = 0; jd < 4; ++jd) o[jd] = wmma_f16(pf[s], vf[s][jd], o[jd]);
}

// ---------------------------------------------------------------------------
// Flash attention (no scaling, per reference).
// Grid: (S/128, H). 256 threads = 8 waves, each wave owns 16 query rows.
// ---------------------------------------------------------------------------
__global__ void __launch_bounds__(256)
flash_attn_kernel(const _Float16* __restrict__ Q, const _Float16* __restrict__ K,
                  const _Float16* __restrict__ Vt, _Float16* __restrict__ Aout) {
  const int lane = threadIdx.x & 31;
  const int wave = threadIdx.x >> 5;
  const int ln = lane & 15, lh = lane >> 4;
  const int h = blockIdx.y;
  const int q0row = (blockIdx.x * 8 + wave) * 16;

  const _Float16* Qh = Q  + h * (S_LEN * D_HEAD);
  const _Float16* Kh = K  + h * (S_LEN * D_HEAD);
  const _Float16* Vh = Vt + h * (D_HEAD * S_LEN);

  __shared__ _Float16 plds[8 * 16 * 64];
  _Float16* pw = plds + wave * (16 * 64);

  // Q fragments (A layout): row = q0row+ln
  const _Float16* qrow = Qh + (q0row + ln) * D_HEAD;
  v16h qf[2];
#pragma unroll
  for (int s = 0; s < 2; ++s) {
    qf[s] = *(const v16h*)(qrow + s * 32 + lh * 16);
  }

  const v8f z8 = {0.f,0.f,0.f,0.f,0.f,0.f,0.f,0.f};
  v8f o[4];
#pragma unroll
  for (int j = 0; j < 4; ++j) o[j] = z8;
  float rm[8], rl[8];
#pragma unroll
  for (int r = 0; r < 8; ++r) { rm[r] = -3.0e38f; rl[r] = 0.f; }

  v16h kfA[2][4], kfB[2][4];
  load_kfrags(Kh, 0, ln, lh, kfA);

  // Double-stepped, double-buffered key loop (S/64 = 64 blocks, even count).
  for (int kb = 0; kb < S_LEN; kb += 128) {
    attn_block(kb, kb + 64, Kh, Vh, pw, ln, lh, qf, kfA, kfB, o, rm, rl);
    const int nb = (kb + 128 < S_LEN) ? kb + 128 : 0;  // clamped redundant preload
    attn_block(kb + 64, nb, Kh, Vh, pw, ln, lh, qf, kfB, kfA, o, rm, rl);
  }

  // ---- normalize + store [H][S][64] ----
  _Float16* Ao = Aout + h * (S_LEN * D_HEAD);
#pragma unroll
  for (int jd = 0; jd < 4; ++jd)
#pragma unroll
    for (int r = 0; r < 8; ++r) {
      float v = o[jd][r] / rl[r];
      Ao[(q0row + r + lh * 8) * D_HEAD + jd * 16 + ln] = (_Float16)v;
    }
}

// ---------------------------------------------------------------------------
// Launch. Workspace layout (bytes):
//   xh 0..8M | wq 8M | wk 10M | wv 12M | wo 14M | Q 16M | K 24M | Vt 32M | A 40M
// ---------------------------------------------------------------------------
extern "C" void kernel_launch(void* const* d_in, const int* in_sizes, int n_in,
                              void* d_out, int out_size, void* d_ws, size_t ws_size,
                              hipStream_t stream) {
  const float* x  = (const float*)d_in[0];
  const float* wq = (const float*)d_in[1];
  const float* wk = (const float*)d_in[2];
  const float* wv = (const float*)d_in[3];
  const float* wo = (const float*)d_in[4];
  const float* bo = (const float*)d_in[5];

  char* ws = (char*)d_ws;
  _Float16* xh  = (_Float16*)(ws);
  _Float16* wqh = (_Float16*)(ws + (size_t)( 8u << 20));
  _Float16* wkh = (_Float16*)(ws + (size_t)(10u << 20));
  _Float16* wvh = (_Float16*)(ws + (size_t)(12u << 20));
  _Float16* woh = (_Float16*)(ws + (size_t)(14u << 20));
  _Float16* Qh  = (_Float16*)(ws + (size_t)(16u << 20));
  _Float16* Kh  = (_Float16*)(ws + (size_t)(24u << 20));
  _Float16* Vth = (_Float16*)(ws + (size_t)(32u << 20));
  _Float16* Ah  = (_Float16*)(ws + (size_t)(40u << 20));

  const int nx = S_LEN * E_DIM;  // 4 Mi elements
  const int nw = E_DIM * E_DIM;  // 1 Mi elements
  f32_to_f16_kernel<<<dim3(nx / 1024), 256, 0, stream>>>(x,  xh,  nx);
  f32_to_f16_kernel<<<dim3(nw / 1024), 256, 0, stream>>>(wq, wqh, nw);
  f32_to_f16_kernel<<<dim3(nw / 1024), 256, 0, stream>>>(wk, wkh, nw);
  f32_to_f16_kernel<<<dim3(nw / 1024), 256, 0, stream>>>(wv, wvh, nw);
  f32_to_f16_kernel<<<dim3(nw / 1024), 256, 0, stream>>>(wo, woh, nw);

  dim3 ggrid(E_DIM / 256, S_LEN / 128);  // (4, 32)
  gemm_xwt_kernel<<<ggrid, 256, 0, stream>>>(xh, wqh, Qh,  nullptr, nullptr, 0);
  gemm_xwt_kernel<<<ggrid, 256, 0, stream>>>(xh, wkh, Kh,  nullptr, nullptr, 0);
  gemm_xwt_kernel<<<ggrid, 256, 0, stream>>>(xh, wvh, Vth, nullptr, nullptr, 1);

  flash_attn_kernel<<<dim3(S_LEN / 128, H_NUM), 256, 0, stream>>>(Qh, Kh, Vth, Ah);

  gemm_xwt_kernel<<<ggrid, 256, 0, stream>>>(Ah, woh, nullptr, bo, (float*)d_out, 2);
}
